// Self_attention_59777354826463
// MI455X (gfx1250) — matrix operand order repaired and checked by reading, not compiled, not run on hardware
//
#include <hip/hip_runtime.h>
#include <hip/hip_bf16.h>
#include <math.h>

// Shapes (fixed by the reference)
#define B_    16
#define S_    1024
#define D_    512
#define PRED_ 256
#define MAXS_ 2048

typedef __attribute__((ext_vector_type(16))) __bf16 v16bf;
typedef __attribute__((ext_vector_type(8)))  __bf16 v8bf;
typedef __attribute__((ext_vector_type(8)))  float  v8f;

union V16 { v16bf v; v8bf h[2]; };

__device__ __forceinline__ v8f wmma_bf16(v16bf a, v16bf b, v8f c) {
  // D = A(16x32 bf16) * B(32x16 bf16) + C(16x16 f32)
  return __builtin_amdgcn_wmma_f32_16x16x32_bf16(
      /*neg_a=*/false, a, /*neg_b=*/false, b,
      /*c_mod=*/(short)0, c, /*reuse_a=*/false, /*reuse_b=*/false);
}

// ---------------------------------------------------------------------------
// fp32 -> bf16 elementwise convert (n must be a multiple of 4)
// ---------------------------------------------------------------------------
__global__ void cvt_bf16_kernel(const float* __restrict__ src,
                                __bf16* __restrict__ dst, int n) {
  int i = (blockIdx.x * blockDim.x + threadIdx.x) * 4;
  if (i + 3 < n) {
    float4 f = *(const float4*)(src + i);
    dst[i + 0] = (__bf16)f.x;
    dst[i + 1] = (__bf16)f.y;
    dst[i + 2] = (__bf16)f.z;
    dst[i + 3] = (__bf16)f.w;
  }
}

// ---------------------------------------------------------------------------
// Tiled transpose-convert: src [B,S,D] f32 -> dst [B,D,S] bf16
// block (32,8), 32x32 tile per block
// ---------------------------------------------------------------------------
__global__ void transpose_bf16_kernel(const float* __restrict__ src,
                                      __bf16* __restrict__ dst) {
  __shared__ __bf16 tile[32][33];
  const int b  = blockIdx.z;
  const int d0 = blockIdx.x * 32;
  const int s0 = blockIdx.y * 32;
  const int tx = threadIdx.x;   // 0..31
  const int ty = threadIdx.y;   // 0..7

  const float* sp = src + ((size_t)b * S_ + s0) * D_ + d0;
  #pragma unroll
  for (int k = 0; k < 4; ++k) {
    int r = ty + k * 8;
    tile[r][tx] = (__bf16)sp[(size_t)r * D_ + tx];   // tile[s][d]
  }
  __syncthreads();
  __bf16* dp = dst + ((size_t)b * D_ + d0) * S_ + s0;
  #pragma unroll
  for (int k = 0; k < 4; ++k) {
    int r = ty + k * 8;
    dp[(size_t)r * S_ + tx] = tile[tx][r];           // dst[d][s]
  }
}

// ---------------------------------------------------------------------------
// Flash attention: one 16-row Q tile per 128-thread block (4 waves).
// All waves compute the 16x32 score tile (redundantly, compute is cheap);
// wave w accumulates PV for D slice [w*128, w*128+128).
// dataB: [B,S,D] bf16 (row major), dataT: [B,D,S] bf16, attnB: [B,S,D] bf16 out
// ---------------------------------------------------------------------------
__global__ void __launch_bounds__(128)
flash_attn_kernel(const __bf16* __restrict__ dataB,
                  const __bf16* __restrict__ dataT,
                  __bf16* __restrict__ attnB) {
  __shared__ __bf16 ldsP[4][16][32];   // per-wave P-tile staging (C->A transpose)

  const int lane = threadIdx.x & 31;
  const int wave = threadIdx.x >> 5;
  const int row  = lane & 15;          // A-frag row / B-frag column / C col
  const int sel  = lane >> 4;          // lane half selector
  const int m0   = blockIdx.x * 16;    // Q-tile base row
  const int b    = blockIdx.y;         // batch
  const int dbase = wave * 128;        // this wave's D slice for PV
  const float scale = 0.044194173824159216f;  // 1/sqrt(512)

  // Preload Q fragments for all 16 K-steps (16x32 bf16 each)
  V16 qf[16];
  const __bf16* qrow = dataB + ((size_t)b * S_ + m0 + row) * D_;
  #pragma unroll
  for (int kk = 0; kk < 16; ++kk) {
    qf[kk].h[0] = *(const v8bf*)(qrow + kk * 32 + sel * 16);
    qf[kk].h[1] = *(const v8bf*)(qrow + kk * 32 + sel * 16 + 8);
  }

  v8f attn[8] = {};                    // 16 x 128 fp32 accumulator (this wave)
  float mrow[8], lrow[8];
  #pragma unroll
  for (int v = 0; v < 8; ++v) { mrow[v] = -INFINITY; lrow[v] = 0.0f; }

  for (int t0 = 0; t0 < S_; t0 += 32) {
    // ---- scores: Q(16x512) . K^T -> 16x32 tile ----
    v8f s0 = {}, s1 = {};
    const __bf16* kp0 = dataB + ((size_t)b * S_ + t0 + row) * D_ + sel * 16;
    const __bf16* kp1 = kp0 + 16 * D_;
    #pragma unroll
    for (int kk = 0; kk < 16; ++kk) {
      v16bf b0 = *(const v16bf*)(kp0 + kk * 32);
      v16bf b1 = *(const v16bf*)(kp1 + kk * 32);
      s0 = wmma_bf16(qf[kk].v, b0, s0);
      s1 = wmma_bf16(qf[kk].v, b1, s1);
    }

    // ---- online softmax over the 32 new columns ----
    #pragma unroll
    for (int v = 0; v < 8; ++v) {
      float a0 = s0[v] * scale;
      float a1 = s1[v] * scale;
      float r = fmaxf(a0, a1);
      r = fmaxf(r, __shfl_xor(r, 1, 32));
      r = fmaxf(r, __shfl_xor(r, 2, 32));
      r = fmaxf(r, __shfl_xor(r, 4, 32));
      r = fmaxf(r, __shfl_xor(r, 8, 32));   // row max within 16-lane half
      float mnew = fmaxf(mrow[v], r);
      float corr = __expf(mrow[v] - mnew);
      float p0 = __expf(a0 - mnew);
      float p1 = __expf(a1 - mnew);
      float su = p0 + p1;
      su += __shfl_xor(su, 1, 32);
      su += __shfl_xor(su, 2, 32);
      su += __shfl_xor(su, 4, 32);
      su += __shfl_xor(su, 8, 32);
      lrow[v] = lrow[v] * corr + su;
      mrow[v] = mnew;
      #pragma unroll
      for (int c = 0; c < 8; ++c) attn[c][v] *= corr;
      // stage P tile (C layout: row M = v + sel*8, cols = lane&15 / +16)
      ldsP[wave][v + sel * 8][row]      = (__bf16)p0;
      ldsP[wave][v + sel * 8][row + 16] = (__bf16)p1;
    }

    // per-wave LDS region; DS ops are in-order per wave, just drain the count
    asm volatile("s_wait_dscnt 0" ::: "memory");

    // ---- read P back as A fragment (16x32 bf16) ----
    V16 pa;
    pa.h[0] = *(const v8bf*)&ldsP[wave][row][sel * 16];
    pa.h[1] = *(const v8bf*)&ldsP[wave][row][sel * 16 + 8];

    // ---- PV: P(16x32) . V(32x128 slice), B-frags contiguous from dataT ----
    const __bf16* vp = dataT + (size_t)b * D_ * S_
                     + (size_t)(dbase + row) * S_ + t0 + sel * 16;
    #pragma unroll
    for (int c = 0; c < 8; ++c) {
      v16bf bv = *(const v16bf*)(vp + (size_t)(c * 16) * S_);
      attn[c] = wmma_bf16(pa.v, bv, attn[c]);
    }
  }

  // ---- normalize and emit attn tile as bf16 ----
  #pragma unroll
  for (int v = 0; v < 8; ++v) {
    float rl = 1.0f / lrow[v];
    int rg = m0 + v + sel * 8;
    __bf16* op = attnB + ((size_t)b * S_ + rg) * D_ + dbase + row;
    #pragma unroll
    for (int c = 0; c < 8; ++c) op[c * 16] = (__bf16)(attn[c][v] * rl);
  }
}

// ---------------------------------------------------------------------------
// Linear: out[:, :S, :] = attn(bf16) @ W^T + bias, fp32 out.
// 128-thread block; wave w computes a 16(M) x 64(N) tile.
// ---------------------------------------------------------------------------
__global__ void __launch_bounds__(128)
linear_kernel(const __bf16* __restrict__ attnB,   // [B*S, D]
              const __bf16* __restrict__ Wb,      // [PRED, D]
              const float*  __restrict__ bias,    // [PRED]
              float* __restrict__ out) {          // [B, MAXS, PRED]
  const int lane  = threadIdx.x & 31;
  const int wave  = threadIdx.x >> 5;
  const int row   = lane & 15;
  const int sel   = lane >> 4;
  const int mrow0 = blockIdx.x * 16;   // flat (b*S + s) tile base
  const int n0    = wave * 64;

  v8f acc[4] = {};
  const __bf16* arow = attnB + (size_t)(mrow0 + row) * D_;
  #pragma unroll
  for (int kk = 0; kk < 16; ++kk) {
    V16 af;
    af.h[0] = *(const v8bf*)(arow + kk * 32 + sel * 16);
    af.h[1] = *(const v8bf*)(arow + kk * 32 + sel * 16 + 8);
    #pragma unroll
    for (int j = 0; j < 4; ++j) {
      const __bf16* wp = Wb + (size_t)(n0 + j * 16 + row) * D_ + kk * 32 + sel * 16;
      v16bf bw = *(const v16bf*)wp;
      acc[j] = wmma_bf16(af.v, bw, acc[j]);
    }
  }

  const int bIdx = mrow0 / S_;         // 16 | S_, so whole tile in one batch
  const int srow = mrow0 % S_;
  #pragma unroll
  for (int j = 0; j < 4; ++j) {
    int col = n0 + j * 16 + row;
    float bv = bias[col];
    float* op = out + ((size_t)bIdx * MAXS_ + srow + sel * 8) * PRED_ + col;
    #pragma unroll
    for (int v = 0; v < 8; ++v) op[(size_t)v * PRED_] = acc[j][v] + bv;
  }
}

// ---------------------------------------------------------------------------
// Zero-fill out[:, S:MAXS, :]
// ---------------------------------------------------------------------------
__global__ void zero_tail_kernel(float* __restrict__ out) {
  size_t i = ((size_t)blockIdx.x * blockDim.x + threadIdx.x) * 4;
  const size_t per_batch = (size_t)S_ * PRED_;         // tail floats per batch
  size_t bIdx = i / per_batch;
  size_t r    = i % per_batch;
  float4 z = {0.f, 0.f, 0.f, 0.f};
  *(float4*)(out + bIdx * (size_t)MAXS_ * PRED_ + per_batch + r) = z;
}

// ---------------------------------------------------------------------------
extern "C" void kernel_launch(void* const* d_in, const int* in_sizes, int n_in,
                              void* d_out, int out_size, void* d_ws, size_t ws_size,
                              hipStream_t stream) {
  (void)in_sizes; (void)n_in; (void)out_size; (void)ws_size;
  const float* data = (const float*)d_in[0];   // [B,S,D]
  const float* W    = (const float*)d_in[1];   // [PRED,D]
  const float* bias = (const float*)d_in[2];   // [PRED]
  float* out = (float*)d_out;                  // [B,MAXS,PRED]

  char* ws = (char*)d_ws;
  __bf16* dataB = (__bf16*)(ws);                                        // 16 MB
  __bf16* dataT = (__bf16*)(ws + (size_t)16 * 1024 * 1024);             // 16 MB
  __bf16* Wb    = (__bf16*)(ws + (size_t)32 * 1024 * 1024);             // 256 KB
  __bf16* attnB = (__bf16*)(ws + (size_t)32 * 1024 * 1024 + 512 * 1024);// 16 MB

  const int nData = B_ * S_ * D_;        // 8388608
  const int nW    = PRED_ * D_;          // 131072

  cvt_bf16_kernel<<<nData / 4 / 256, 256, 0, stream>>>(data, dataB, nData);
  cvt_bf16_kernel<<<nW / 4 / 256, 256, 0, stream>>>(W, Wb, nW);
  transpose_bf16_kernel<<<dim3(D_ / 32, S_ / 32, B_), dim3(32, 8), 0, stream>>>(data, dataT);
  flash_attn_kernel<<<dim3(S_ / 16, B_), 128, 0, stream>>>(dataB, dataT, attnB);
  linear_kernel<<<(B_ * S_) / 16, 128, 0, stream>>>(attnB, Wb, bias, out);
  zero_tail_kernel<<<(B_ * S_ * PRED_ / 4) / 256, 256, 0, stream>>>(out);
}
